// MambaBlock_4664334483708
// MI455X (gfx1250) — compile-verified, matrix operand order checked by reading, not run on hardware
//
#include <hip/hip_runtime.h>

#define USE_TDM 1

// ---------------- model dims ----------------
#define D_MODEL 1024
#define D_INNER 2048
#define D_STATE 16
#define DT_RANK 64
#define D_CONV  4
#define BB      2
#define LL      2048
#define M_ROWS  (BB*LL)          // 4096
#define N_SMALL (DT_RANK + 2*D_STATE)  // 96

typedef __attribute__((ext_vector_type(16))) __bf16          bf16x16;
typedef __attribute__((ext_vector_type(8)))  float           f32x8;
typedef __attribute__((ext_vector_type(8)))  unsigned short  u16x8;
typedef __attribute__((ext_vector_type(4)))  unsigned int    u32x4;
typedef __attribute__((ext_vector_type(8)))  unsigned int    u32x8;

__device__ __forceinline__ unsigned short f32_to_bf16(float f) {
    unsigned int u = __float_as_uint(f);
    unsigned int lsb = (u >> 16) & 1u;
    u += 0x7fffu + lsb;            // round-to-nearest-even
    return (unsigned short)(u >> 16);
}

// ---------------------------------------------------------------------------
// Weight packing: W is (N x K) row-major fp32.  Produce bf16 "B-operand" tiles:
// for each (kt, nt) 32x16 tile, lane l holds 16 elements with
//   n = nt*16 + (l & 15)
//   k(e) = kt*32 + (e/8)*16 + (l/16)*8 + (e%8)        e = 0..15
// stored contiguously: P[(((kt*NT)+nt)*32 + l)*16 + e]
// => for a fixed kt, NW consecutive n-tiles are one contiguous NW*1024B chunk.
// ---------------------------------------------------------------------------
__global__ void pack_weight_kernel(const float* __restrict__ W,
                                   unsigned short* __restrict__ P,
                                   int N, int K) {
    size_t idx = (size_t)blockIdx.x * 256 + threadIdx.x;
    if (idx >= (size_t)N * K) return;
    int e    = (int)(idx & 15);
    int lane = (int)((idx >> 4) & 31);
    size_t t = idx >> 9;
    int NT   = N >> 4;
    int nt   = (int)(t % NT);
    int kt   = (int)(t / NT);
    int n = nt * 16 + (lane & 15);
    int k = kt * 32 + ((e >> 3) << 4) + ((lane >> 4) << 3) + (e & 7);
    P[idx] = f32_to_bf16(W[(size_t)n * K + k]);
}

// Fused [wd1 (64 rows) | wb (16) | wc (16)] as one (96 x 2048) weight.
__global__ void pack_small_kernel(const float* __restrict__ wd1,
                                  const float* __restrict__ wb,
                                  const float* __restrict__ wc,
                                  unsigned short* __restrict__ P) {
    size_t idx = (size_t)blockIdx.x * 256 + threadIdx.x;
    if (idx >= (size_t)N_SMALL * D_INNER) return;
    int e    = (int)(idx & 15);
    int lane = (int)((idx >> 4) & 31);
    size_t t = idx >> 9;
    int NT   = N_SMALL >> 4;                 // 6
    int nt   = (int)(t % NT);
    int kt   = (int)(t / NT);
    int n = nt * 16 + (lane & 15);
    int k = kt * 32 + ((e >> 3) << 4) + ((lane >> 4) << 3) + (e & 7);
    const float* W; int nr;
    if (n < 64)      { W = wd1; nr = n; }
    else if (n < 80) { W = wb;  nr = n - 64; }
    else             { W = wc;  nr = n - 80; }
    P[idx] = f32_to_bf16(W[(size_t)nr * D_INNER + k]);
}

__global__ void a_neg_kernel(const float* __restrict__ A_log, float* __restrict__ A_neg) {
    int i = blockIdx.x * 256 + threadIdx.x;
    if (i < D_INNER * D_STATE) A_neg[i] = -__expf(A_log[i]);
}

// ---------------------------------------------------------------------------
// RMSNorm: one block per (b,l) row of 1024; output bf16 for WMMA A operand.
// ---------------------------------------------------------------------------
__global__ __launch_bounds__(256) void rmsnorm_kernel(const float* __restrict__ resid,
                                                      const float* __restrict__ norm_w,
                                                      unsigned short* __restrict__ xnorm) {
    int row = blockIdx.x;
    const float* xr = resid + (size_t)row * D_MODEL;
    float ss = 0.f;
    for (int i = threadIdx.x; i < D_MODEL; i += 256) { float v = xr[i]; ss += v * v; }
    #pragma unroll
    for (int off = 16; off > 0; off >>= 1) ss += __shfl_down(ss, off, 32);
    __shared__ float red[8];
    int wv = threadIdx.x >> 5, ln = threadIdx.x & 31;
    if (ln == 0) red[wv] = ss;
    __syncthreads();
    if (threadIdx.x == 0) {
        float t = 0.f;
        #pragma unroll
        for (int i = 0; i < 8; i++) t += red[i];
        red[0] = rsqrtf(t / (float)D_MODEL + 1e-5f);
    }
    __syncthreads();
    float scale = red[0];
    for (int i = threadIdx.x; i < D_MODEL; i += 256)
        xnorm[(size_t)row * D_MODEL + i] = f32_to_bf16(xr[i] * scale * norm_w[i]);
}

// ---------------------------------------------------------------------------
// bf16 WMMA GEMM with LDS-staged B operand (TDM double-buffered DMA).
// Block = 8 waves = 128 M rows x (16*NW) N cols.  All waves share the B chunk,
// which is staged once per k-step into LDS (tensor_load_to_lds on wave 0).
// mode 0: plain (optional bf16 mirror)   mode 1: +bias[col], softplus
// mode 2: +extra[row*N+col] (residual add)
// ---------------------------------------------------------------------------
template<int NW>
__global__ __launch_bounds__(256) void gemm_bf16_kernel(
        const unsigned short* __restrict__ A, int lda,
        const unsigned short* __restrict__ Bp,
        float* __restrict__ C, unsigned short* __restrict__ Cbf,
        const float* __restrict__ extra,
        int N, int K, int mode) {
    const int lane = threadIdx.x & 31;
    const int wave = threadIdx.x >> 5;
    const int mt = blockIdx.y * 8 + wave;          // 16-row tile index
    const int ntBase = blockIdx.x * NW;            // first 16-col tile
    const int NT = N >> 4;
    const int nk = K >> 5;
    constexpr int CHUNK_ELEMS = NW * 512;          // bf16 elems per k-step chunk
    __shared__ __align__(64) unsigned short sB[2][CHUNK_ELEMS];

    auto stage = [&](int kt, int buf) {
        const unsigned short* gsrc = Bp + ((size_t)kt * NT + ntBase) * 512;
#if USE_TDM
        if (threadIdx.x < 32) {                    // wave 0 issues the DMA
            unsigned long long ga = (unsigned long long)(size_t)gsrc;
            unsigned int ldsOff = (unsigned int)(size_t)(&sB[buf][0]); // flat LDS addr[31:0] = LDS offset
            unsigned int te = (unsigned int)CHUNK_ELEMS;
            u32x4 g0;
            g0[0] = 1u;                                            // count=1, valid user D#
            g0[1] = ldsOff;                                        // lds_addr
            g0[2] = (unsigned int)ga;                              // global_addr[31:0]
            g0[3] = (unsigned int)(ga >> 32) | 0x80000000u;        // global_addr[56:32] | type=2
            u32x8 g1;
            g1[0] = 0x00010000u;                                   // data_size=1 (2B), no mask/pad
            g1[1] = (te & 0xffffu) << 16;                          // tensor_dim0[15:0]
            g1[2] = ((te >> 16) & 0xffffu) | (1u << 16);           // tensor_dim0[31:16] | tensor_dim1=1
            g1[3] = (te & 0xffffu) << 16;                          // tile_dim0 = chunk elems
            g1[4] = 1u;                                            // tile_dim1=1, tile_dim2=0
            g1[5] = te;                                            // tensor_dim0_stride[31:0]
            g1[6] = 0u;
            g1[7] = 0u;
            asm volatile("tensor_load_to_lds %0, %1" :: "s"(g0), "s"(g1) : "memory");
        }
#else
        for (int i = threadIdx.x; i * 8 < CHUNK_ELEMS; i += 256)
            ((u16x8*)&sB[buf][0])[i] = ((const u16x8*)gsrc)[i];
#endif
    };

    stage(0, 0);
#if USE_TDM
    if (threadIdx.x < 32) __builtin_amdgcn_s_wait_tensorcnt(0);
#endif
    __syncthreads();

    f32x8 acc[NW];
    #pragma unroll
    for (int j = 0; j < NW; j++) acc[j] = (f32x8)(0.0f);

    const unsigned short* arow =
        A + (size_t)(mt * 16 + (lane & 15)) * lda + ((lane >> 4) << 3);

    for (int kt = 0; kt < nk; kt++) {
        const int buf = kt & 1;
        if (kt + 1 < nk) stage(kt + 1, buf ^ 1);   // prefetch next chunk via TDM

        union { u16x8 h[2]; bf16x16 v; } af;
        af.h[0] = *(const u16x8*)(arow);
        af.h[1] = *(const u16x8*)(arow + 16);
        arow += 32;
        __builtin_prefetch(arow, 0, 1);            // global_prefetch_b8 (A stream)

        #pragma unroll
        for (int j = 0; j < NW; j++) {
            bf16x16 bf = *(const bf16x16*)&sB[buf][(j * 32 + lane) * 16];
            acc[j] = __builtin_amdgcn_wmma_f32_16x16x32_bf16(
                false, af.v, false, bf, (short)0, acc[j], false, false);
        }

#if USE_TDM
        if (threadIdx.x < 32) __builtin_amdgcn_s_wait_tensorcnt(0);
#endif
        __syncthreads();
    }

    const int colLane = lane & 15;
    const int rowOff  = (lane >> 4) * 8;
    #pragma unroll
    for (int j = 0; j < NW; j++) {
        int col = (ntBase + j) * 16 + colLane;
        #pragma unroll
        for (int r = 0; r < 8; r++) {
            int row = mt * 16 + rowOff + r;
            float v = acc[j][r];
            size_t ci = (size_t)row * N + col;
            if (mode == 1) {                        // bias + softplus
                v += extra[col];
                v = (v > 20.f) ? v : log1pf(__expf(v));
            } else if (mode == 2) {                 // residual add
                v += extra[ci];
            }
            C[ci] = v;
            if (Cbf) Cbf[ci] = f32_to_bf16(v);
        }
    }
}

// ---------------------------------------------------------------------------
// Causal depthwise conv (K=4) + SiLU. One thread per (b,l,e).
// ---------------------------------------------------------------------------
__global__ __launch_bounds__(256) void conv_silu_kernel(
        const float* __restrict__ xin, const float* __restrict__ conv_w,
        const float* __restrict__ conv_b,
        float* __restrict__ xc, unsigned short* __restrict__ xcbf) {
    size_t idx = (size_t)blockIdx.x * 256 + threadIdx.x;
    if (idx >= (size_t)M_ROWS * D_INNER) return;
    int e = (int)(idx % D_INNER);
    long long bl = (long long)(idx / D_INNER);
    int l = (int)(bl % LL);
    float acc = conv_b[e];
    #pragma unroll
    for (int k = 0; k < D_CONV; k++) {
        int ls = l + k - (D_CONV - 1);
        float xv = (ls >= 0) ? xin[(bl + (k - (D_CONV - 1))) * D_INNER + e] : 0.f;
        acc += xv * conv_w[e * D_CONV + k];
    }
    float s = acc / (1.f + __expf(-acc));
    xc[idx] = s;
    xcbf[idx] = f32_to_bf16(s);
}

// ---------------------------------------------------------------------------
// Selective scan: lane = one (b,e,n) state. 16-lane shfl_xor reduce for y.
// B/C rows staged to LDS in chunks. Emits (y + x*D)*silu(skip) in bf16.
// ---------------------------------------------------------------------------
#define SCAN_CH 8
__global__ __launch_bounds__(256) void scan_kernel(
        const float* __restrict__ xc, const float* __restrict__ delta,
        const float* __restrict__ smallf /* (4096 x 96): [d1|B|C] */,
        const float* __restrict__ A_neg, const float* __restrict__ W_D,
        const float* __restrict__ skip, unsigned short* __restrict__ ybf) {
    int b  = blockIdx.x >> 7;                    // 128 blocks per batch (2048/16)
    int e0 = (blockIdx.x & 127) << 4;
    int g  = threadIdx.x >> 4;                   // channel group within block
    int n  = threadIdx.x & 15;                   // state index
    int e  = e0 + g;

    float A_en = A_neg[e * D_STATE + n];
    float wd   = W_D[e];
    float h = 0.f;

    __shared__ float sBC[SCAN_CH][32];
    for (int l0 = 0; l0 < LL; l0 += SCAN_CH) {
        for (int i = threadIdx.x; i < SCAN_CH * 32; i += 256) {
            int s = i >> 5, c = i & 31;
            sBC[s][c] = smallf[((size_t)(b * LL + l0 + s)) * N_SMALL + DT_RANK + c];
        }
        __syncthreads();
        #pragma unroll
        for (int s = 0; s < SCAN_CH; s++) {
            size_t idx = ((size_t)(b * LL + l0 + s)) * D_INNER + e;
            float dt = delta[idx];
            float xt = xc[idx];
            h = __expf(dt * A_en) * h + dt * xt * sBC[s][n];
            float y = h * sBC[s][16 + n];
            #pragma unroll
            for (int off = 8; off >= 1; off >>= 1) y += __shfl_xor(y, off, 16);
            if (n == 0) {
                float sk = skip[idx];
                float act = (y + xt * wd) * (sk / (1.f + __expf(-sk)));
                ybf[idx] = f32_to_bf16(act);
            }
        }
        __syncthreads();
    }
}

// ---------------------------------------------------------------------------
// Workspace layout (bytes)
// ---------------------------------------------------------------------------
static constexpr size_t OFF_W_IN    = 0;                                    // 2048x1024 bf16
static constexpr size_t OFF_W_SKIP  = OFF_W_IN    + (size_t)D_INNER*D_MODEL*2;
static constexpr size_t OFF_W_OUT   = OFF_W_SKIP  + (size_t)D_INNER*D_MODEL*2;
static constexpr size_t OFF_W_SMALL = OFF_W_OUT   + (size_t)D_MODEL*D_INNER*2;
static constexpr size_t OFF_W_D2    = OFF_W_SMALL + (size_t)N_SMALL*D_INNER*2;
static constexpr size_t OFF_A_NEG   = OFF_W_D2    + (size_t)D_INNER*DT_RANK*2;
static constexpr size_t OFF_XNORM   = OFF_A_NEG   + (size_t)D_INNER*D_STATE*4;
static constexpr size_t OFF_XIN     = OFF_XNORM   + (size_t)M_ROWS*D_MODEL*2;
static constexpr size_t OFF_SKIP    = OFF_XIN     + (size_t)M_ROWS*D_INNER*4;
static constexpr size_t OFF_XC      = OFF_SKIP    + (size_t)M_ROWS*D_INNER*4;
static constexpr size_t OFF_XCBF    = OFF_XC      + (size_t)M_ROWS*D_INNER*4;
static constexpr size_t OFF_SMALL   = OFF_XCBF    + (size_t)M_ROWS*D_INNER*2;
static constexpr size_t OFF_SMALLBF = OFF_SMALL   + (size_t)M_ROWS*N_SMALL*4;
static constexpr size_t OFF_DELTA   = OFF_SMALLBF + (size_t)M_ROWS*N_SMALL*2;
static constexpr size_t OFF_YBF     = OFF_DELTA   + (size_t)M_ROWS*D_INNER*4;

extern "C" void kernel_launch(void* const* d_in, const int* in_sizes, int n_in,
                              void* d_out, int out_size, void* d_ws, size_t ws_size,
                              hipStream_t stream) {
    const float* resid  = (const float*)d_in[0];
    const float* norm_w = (const float*)d_in[1];
    const float* skip_w = (const float*)d_in[2];
    const float* in_w   = (const float*)d_in[3];
    const float* conv_w = (const float*)d_in[4];
    const float* conv_b = (const float*)d_in[5];
    const float* wd1    = (const float*)d_in[6];
    const float* wd2    = (const float*)d_in[7];
    const float* wd2_b  = (const float*)d_in[8];
    const float* wb     = (const float*)d_in[9];
    const float* wc     = (const float*)d_in[10];
    const float* A_log  = (const float*)d_in[11];
    const float* W_D    = (const float*)d_in[12];
    const float* out_w  = (const float*)d_in[13];
    float* out = (float*)d_out;

    char* ws = (char*)d_ws;
    unsigned short* wInP   = (unsigned short*)(ws + OFF_W_IN);
    unsigned short* wSkipP = (unsigned short*)(ws + OFF_W_SKIP);
    unsigned short* wOutP  = (unsigned short*)(ws + OFF_W_OUT);
    unsigned short* wSmallP= (unsigned short*)(ws + OFF_W_SMALL);
    unsigned short* wD2P   = (unsigned short*)(ws + OFF_W_D2);
    float*          aNeg   = (float*)(ws + OFF_A_NEG);
    unsigned short* xnorm  = (unsigned short*)(ws + OFF_XNORM);
    float*          xin    = (float*)(ws + OFF_XIN);
    float*          skip   = (float*)(ws + OFF_SKIP);
    float*          xc     = (float*)(ws + OFF_XC);
    unsigned short* xcbf   = (unsigned short*)(ws + OFF_XCBF);
    float*          smallf = (float*)(ws + OFF_SMALL);
    unsigned short* smallbf= (unsigned short*)(ws + OFF_SMALLBF);
    float*          deltaf = (float*)(ws + OFF_DELTA);
    unsigned short* ybf    = (unsigned short*)(ws + OFF_YBF);

    // 1) Weight prep (bf16, WMMA B-operand layout) + A = -exp(A_log)
    pack_weight_kernel<<<(D_INNER*D_MODEL+255)/256, 256, 0, stream>>>(in_w,   wInP,   D_INNER, D_MODEL);
    pack_weight_kernel<<<(D_INNER*D_MODEL+255)/256, 256, 0, stream>>>(skip_w, wSkipP, D_INNER, D_MODEL);
    pack_weight_kernel<<<(D_MODEL*D_INNER+255)/256, 256, 0, stream>>>(out_w,  wOutP,  D_MODEL, D_INNER);
    pack_weight_kernel<<<(D_INNER*DT_RANK+255)/256, 256, 0, stream>>>(wd2,    wD2P,   D_INNER, DT_RANK);
    pack_small_kernel <<<(N_SMALL*D_INNER+255)/256, 256, 0, stream>>>(wd1, wb, wc, wSmallP);
    a_neg_kernel      <<<(D_INNER*D_STATE+255)/256, 256, 0, stream>>>(A_log, aNeg);

    // 2) RMSNorm -> bf16
    rmsnorm_kernel<<<M_ROWS, 256, 0, stream>>>(resid, norm_w, xnorm);

    // 3) xin = xnorm @ in_w^T ; skip = xnorm @ skip_w^T   (WMMA + TDM)
    gemm_bf16_kernel<4><<<dim3(D_INNER/64, M_ROWS/128), 256, 0, stream>>>(
        xnorm, D_MODEL, wInP,   xin,  nullptr, nullptr, D_INNER, D_MODEL, 0);
    gemm_bf16_kernel<4><<<dim3(D_INNER/64, M_ROWS/128), 256, 0, stream>>>(
        xnorm, D_MODEL, wSkipP, skip, nullptr, nullptr, D_INNER, D_MODEL, 0);

    // 4) causal depthwise conv + SiLU
    conv_silu_kernel<<<(size_t)M_ROWS*D_INNER/256, 256, 0, stream>>>(xin, conv_w, conv_b, xc, xcbf);

    // 5) fused [delta1 | B | C] = xc @ [wd1|wb|wc]^T   (N=96, WMMA + TDM)
    gemm_bf16_kernel<2><<<dim3(N_SMALL/32, M_ROWS/128), 256, 0, stream>>>(
        xcbf, D_INNER, wSmallP, smallf, smallbf, nullptr, N_SMALL, D_INNER, 0);

    // 6) delta = softplus(delta1 @ wd2^T + b)   (K=64, WMMA + TDM)
    gemm_bf16_kernel<4><<<dim3(D_INNER/64, M_ROWS/128), 256, 0, stream>>>(
        smallbf, N_SMALL, wD2P, deltaf, nullptr, wd2_b, D_INNER, DT_RANK, 1);

    // 7) selective scan + gating -> y_bf16
    scan_kernel<<<BB*(D_INNER/16), 256, 0, stream>>>(xc, deltaf, smallf, aNeg, W_D, skip, ybf);

    // 8) out = resid + y @ out_w^T   (WMMA + TDM, residual-add epilogue)
    gemm_bf16_kernel<4><<<dim3(D_MODEL/64, M_ROWS/128), 256, 0, stream>>>(
        ybf, D_INNER, wOutP, out, nullptr, resid, D_MODEL, D_INNER, 2);
}
